// MultiHeadAttnC_22419729285212
// MI455X (gfx1250) — compile-verified, matrix operand order checked by reading, not run on hardware
//
#include <hip/hip_runtime.h>

// MultiHeadAttn for MI455X (gfx1250): fused projection + flash attention,
// all matmuls via v_wmma_f32_16x16x32_f16 (wave32).
//
// Shapes: B=8, D=128, L=2048, H=8, DH=16.

#define B_  8
#define D_  128
#define L_  2048
#define H_  8
#define DH_ 16
#define NEGV (-1.0e30f)

typedef __attribute__((ext_vector_type(16))) _Float16 v16h;
typedef __attribute__((ext_vector_type(8)))  _Float16 h8;
typedef __attribute__((ext_vector_type(8)))  float    v8f;
typedef __attribute__((ext_vector_type(4)))  float    f4;

union V16 { v16h v; h8 h[2]; };

__device__ __forceinline__ float rmax16(float v) {
  // reduce across the 16-lane half-group (xor masks < 16 stay in-group)
  v = fmaxf(v, __shfl_xor(v, 8, 32));
  v = fmaxf(v, __shfl_xor(v, 4, 32));
  v = fmaxf(v, __shfl_xor(v, 2, 32));
  v = fmaxf(v, __shfl_xor(v, 1, 32));
  return v;
}
__device__ __forceinline__ float rsum16(float v) {
  v += __shfl_xor(v, 8, 32);
  v += __shfl_xor(v, 4, 32);
  v += __shfl_xor(v, 2, 32);
  v += __shfl_xor(v, 1, 32);
  return v;
}

// ---------------------------------------------------------------------------
// Kernel 1: fused projection  [w_mem; w_query] (384x128) @ queries(b,128,L)
// Writes f16:  K  -> [b][h][l][16]
//              Q  -> [b][h][l][16]   (pre-scaled by DH^-0.5 = 0.25)
//              Vt -> [b][128 rows][l]  (natural GEMM layout == V transposed)
// Grid: B * L/16 blocks of 256 threads (8 waves; each wave does 3 o-tiles).
// ---------------------------------------------------------------------------
__global__ __launch_bounds__(256) void proj_kernel(
    const float* __restrict__ qin, const float* __restrict__ wmem,
    const float* __restrict__ wqry, _Float16* __restrict__ Kf,
    _Float16* __restrict__ Qf, _Float16* __restrict__ Vt)
{
  __shared__ _Float16 Lq[16 * 136];   // queries tile, transposed [l][d], padded

  const int blk   = blockIdx.x;
  const int b     = blk >> 7;          // / (L/16)
  const int ltile = (blk & 127) * 16;
  const int t     = threadIdx.x;

  // Stage queries[b, 0:128, ltile:ltile+16] -> LDS as f16 [l][d]
  {
    const int j  = t & 15;        // l within tile (coalesced along L)
    const int d0 = t >> 4;        // 16 d-rows per round
    const float* src = qin + (size_t)b * D_ * L_ + ltile + j;
#pragma unroll
    for (int r = 0; r < 8; ++r) {
      const int d = d0 + 16 * r;
      Lq[j * 136 + d] = (_Float16)src[(size_t)d * L_];
    }
  }
  __syncthreads();

  const int lane = t & 31;
  // wave id is uniform across the wave -> force it into an SGPR so all the
  // tile-kind branches below become scalar branches (no exec masking; keeps
  // EXEC structurally all-ones around the WMMAs).
  const int w    = __builtin_amdgcn_readfirstlane(t >> 5);
  const int l16  = lane & 15;
  const int koff = (lane & 16) ? 8 : 0;
  const int hi8  = (lane & 16) ? 8 : 0;

  // B operands (shared by this wave's 3 o-tiles): queries chunk, 32 d x 16 l
  V16 bq[4];
#pragma unroll
  for (int kc = 0; kc < 4; ++kc) {
    const _Float16* p = &Lq[l16 * 136 + kc * 32 + koff];
    bq[kc].h[0] = *(const h8*)p;
    bq[kc].h[1] = *(const h8*)(p + 16);
  }

#pragma unroll
  for (int ti = 0; ti < 3; ++ti) {
    const int ot    = w + ti * 8;     // 0..23  (24 row-tiles of the 384 rows)
    const int obase = ot * 16;        // uniform (SGPR): scalar branches below
    // A-matrix row (A layout: lanes 0-15 and 16-31 both carry M = lane&15)
    const float* wbase;
    int arow;
    if (obase < 256) { wbase = wmem; arow = obase + l16; }
    else             { wbase = wqry; arow = obase - 256 + l16; }
    const float* wr = wbase + (size_t)arow * D_;

    v8f acc = {};
#pragma unroll
    for (int kc = 0; kc < 4; ++kc) {
      V16 a;
      const float* p0 = wr + kc * 32 + koff;
      const f4 x0 = *(const f4*)(p0);
      const f4 x1 = *(const f4*)(p0 + 4);
      const f4 y0 = *(const f4*)(p0 + 16);
      const f4 y1 = *(const f4*)(p0 + 20);
#pragma unroll
      for (int e = 0; e < 4; ++e) {
        a.h[0][e]     = (_Float16)x0[e];
        a.h[0][e + 4] = (_Float16)x1[e];
        a.h[1][e]     = (_Float16)y0[e];
        a.h[1][e + 4] = (_Float16)y1[e];
      }
      acc = __builtin_amdgcn_wmma_f32_16x16x32_f16(
          false, a.v, false, bq[kc].v, (short)0, acc, false, false);
    }

    // Scatter C tile (reg r -> row obase+r / obase+r+8, lane -> l column)
    const int l = ltile + l16;
    if (obase < 128) {                       // K rows
      const int hh = obase >> 4;
      _Float16* dst = Kf + ((size_t)(b * H_ + hh) * L_ + l) * DH_;
#pragma unroll
      for (int r = 0; r < 8; ++r) dst[r + hi8] = (_Float16)acc[r];
    } else if (obase < 256) {                // V rows (kept transposed)
      _Float16* dst = Vt + (size_t)(b * 128 + (obase - 128)) * L_ + l;
#pragma unroll
      for (int r = 0; r < 8; ++r) dst[(size_t)(r + hi8) * L_] = (_Float16)acc[r];
    } else {                                 // Q rows, pre-scale by 0.25
      const int hh = (obase - 256) >> 4;
      _Float16* dst = Qf + ((size_t)(b * H_ + hh) * L_ + l) * DH_;
#pragma unroll
      for (int r = 0; r < 8; ++r) dst[r + hi8] = (_Float16)(acc[r] * 0.25f);
    }
  }
}

// ---------------------------------------------------------------------------
// Kernel 2: flash attention. One wave per (b, h, 16-row q-tile), 4 waves/block.
// Streams keys in chunks of 32: 2x QK^T WMMA (K-dim 16 padded to 32), online
// softmax, P transposed via per-wave LDS, 1x fully-utilized K=32 PV WMMA.
// ---------------------------------------------------------------------------
__global__ __launch_bounds__(128) void attn_kernel(
    const _Float16* __restrict__ Kf, const _Float16* __restrict__ Qf,
    const _Float16* __restrict__ Vt, const float* __restrict__ mask,
    float* __restrict__ out)
{
  __shared__ _Float16 Pls[4 * 16 * 48];    // per-wave 16x32 P tile, stride 48

  const int lane = threadIdx.x & 31;
  const int w    = __builtin_amdgcn_readfirstlane(threadIdx.x >> 5);
  const int tile = blockIdx.x * 4 + w;     // uniform (SGPR) -> scalar bases
  const int qt   = tile & 127;             // L/16 = 128 q-tiles
  const int hb   = tile >> 7;
  const int h    = hb & 7;
  const int b    = hb >> 3;
  const int l16  = lane & 15;
  const int koff = (lane & 16) ? 8 : 0;
  const int hi8  = (lane & 16) ? 8 : 0;
  _Float16* P = &Pls[w * 16 * 48];

  const _Float16* Qb = Qf + ((size_t)(b * H_ + h) * L_ + qt * 16) * DH_;
  const _Float16* Kb = Kf + ((size_t)(b * H_ + h) * L_) * DH_;
  const _Float16* Vb = Vt + (size_t)(b * D_ + h * DH_) * L_;
  const float*    mb = mask + (size_t)b * L_;

  const h8 zero = {};
  V16 aq;                                   // Q tile 16x32 (upper K half = 0)
  aq.h[0] = *(const h8*)(Qb + l16 * DH_ + koff);
  aq.h[1] = zero;

  v8f o = {};
  float rm[8], rl[8];
#pragma unroll
  for (int r = 0; r < 8; ++r) { rm[r] = -3.0e38f; rl[r] = 0.0f; }

  for (int kt = 0; kt < L_; kt += 32) {
    V16 bk0, bk1;
    bk0.h[0] = *(const h8*)(Kb + (size_t)(kt + l16) * DH_ + koff);
    bk0.h[1] = zero;
    bk1.h[0] = *(const h8*)(Kb + (size_t)(kt + 16 + l16) * DH_ + koff);
    bk1.h[1] = zero;

    v8f s0 = {}, s1 = {};
    s0 = __builtin_amdgcn_wmma_f32_16x16x32_f16(
        false, aq.v, false, bk0.v, (short)0, s0, false, false);
    s1 = __builtin_amdgcn_wmma_f32_16x16x32_f16(
        false, aq.v, false, bk1.v, (short)0, s1, false, false);

    const float m0 = mb[kt + l16];
    const float m1 = mb[kt + 16 + l16];
    const float n0 = (1.0f - m0) * NEGV;
    const float n1 = (1.0f - m1) * NEGV;

    if (kt + 32 < L_) {   // prefetch next K/V tiles (global_prefetch_b8)
      __builtin_prefetch(Kb + (size_t)(kt + 32 + l16) * DH_, 0, 1);
      __builtin_prefetch(Vb + (size_t)l16 * L_ + kt + 32, 0, 1);
    }

#pragma unroll
    for (int r = 0; r < 8; ++r) {
      const float v0 = s0[r] * m0 + n0;
      const float v1 = s1[r] * m1 + n1;
      // fold both 16-key sub-chunks before the cross-lane reductions
      const float cm = rmax16(fmaxf(v0, v1));
      const float nm = fmaxf(rm[r], cm);
      const float al = __expf(rm[r] - nm);
      const float p0 = __expf(v0 - nm);
      const float p1 = __expf(v1 - nm);
      rl[r] = rl[r] * al + rsum16(p0 + p1);
      rm[r] = nm;
      o[r] *= al;
      const int row = r + hi8;
      P[row * 48 + l16]      = (_Float16)p0;
      P[row * 48 + 16 + l16] = (_Float16)p1;
    }

    // C-layout -> A-layout transpose handoff within the wave (DS in-order,
    // wait for the b16 stores before the b128 reads)
    asm volatile("s_wait_dscnt 0" ::: "memory");

    V16 ap, bv;
    ap.h[0] = *(const h8*)&P[l16 * 48 + koff];
    ap.h[1] = *(const h8*)&P[l16 * 48 + 16 + koff];
    const _Float16* vp = Vb + (size_t)l16 * L_ + kt + koff;
    bv.h[0] = *(const h8*)vp;
    bv.h[1] = *(const h8*)(vp + 16);

    o = __builtin_amdgcn_wmma_f32_16x16x32_f16(
        false, ap.v, false, bv.v, (short)0, o, false, false);
  }

  // Normalize and scatter: out[b][h*16+dh][q]
#pragma unroll
  for (int r = 0; r < 8; ++r) {
    const float inv = 1.0f / rl[r];
    const int q = qt * 16 + r + hi8;
    out[(size_t)(b * D_ + h * DH_ + l16) * L_ + q] = o[r] * inv;
  }
}

// ---------------------------------------------------------------------------
extern "C" void kernel_launch(void* const* d_in, const int* in_sizes, int n_in,
                              void* d_out, int out_size, void* d_ws, size_t ws_size,
                              hipStream_t stream) {
  const float* queries = (const float*)d_in[0];
  const float* mask    = (const float*)d_in[1];
  const float* wmem    = (const float*)d_in[2];
  const float* wqry    = (const float*)d_in[3];
  float* out = (float*)d_out;

  // Workspace: K (4MB) + Q (4MB) + Vt (4MB), all f16. ~12.6 MB total.
  _Float16* Kf = (_Float16*)d_ws;
  _Float16* Qf = Kf + (size_t)B_ * H_ * L_ * DH_;
  _Float16* Vt = Qf + (size_t)B_ * H_ * L_ * DH_;

  proj_kernel<<<B_ * (L_ / 16), 256, 0, stream>>>(queries, wmem, wqry, Kf, Qf, Vt);
  attn_kernel<<<(B_ * H_ * (L_ / 16)) / 4, 128, 0, stream>>>(Kf, Qf, Vt, mask, out);
}
